// FactorizedSpectralConv_48799418417156
// MI455X (gfx1250) — compile-verified, matrix operand order checked by reading, not run on hardware
//
#include <hip/hip_runtime.h>

// ---------------------------------------------------------------------------
// FactorizedSpectralConv on gfx1250 (MI455X): 3 WMMA GEMM stages per axis.
//   branch0: Xf = F @ X ; Y = mix(Xf) ; out  = G @ Y
//   branch1: same with x/y swapped    ; out += G @ Y
// All matrix math: v_wmma_f32_16x16x32_f16 (f16 in, f32 accumulate).
//
// Fragment-ready storage: 16-bit fragments want per-lane K order
//   lanes 0-15 : K = 0..7,16..23   lanes 16-31 : K = 8..15,24..31
// so every 32-K row is stored with K-groups permuted (0,2,1,3); a fragment
// is then two contiguous 16B loads (global_load_b128 / ds_load_b128).
// ---------------------------------------------------------------------------

typedef __attribute__((ext_vector_type(16))) _Float16 v16h;
typedef __attribute__((ext_vector_type(8)))  _Float16 v8h;
typedef __attribute__((ext_vector_type(8)))  float    v8f;

#define Bn 8
#define S 256
#define Cc 64

// K-group permutation (involution): 0->0, 8->16, 16->8, 24->24 (+ low 3 bits)
__device__ __forceinline__ int pos8(int kk8) {            // kk8 in {0,8,16,24}
  return ((kk8 & 8) << 1) | ((kk8 & 16) >> 1);
}
__device__ __forceinline__ int posOf(int k) {             // k in [0,32)
  return pos8(k & 24) + (k & 7);
}

// load 16 contiguous halves (one fragment) as 2 x b128
__device__ __forceinline__ v16h load_frag16(const _Float16* p) {
  const v8h* q = (const v8h*)p;
  v8h lo = q[0], hi = q[1];
  v16h r;
#pragma unroll
  for (int e = 0; e < 8; ++e) { r[e] = lo[e]; r[e + 8] = hi[e]; }
  return r;
}

// ---------------------------------------------------------------------------
// Precompute F_t / G_t in fragment-ready layout.
//   F_t: [m16=4][kc=8][ln=16][32]   F[c=m16*16+ln][t=kc*32+kval(pos)]
//   G_t: [m16=16][kc=2][ln=16][32]  G[x=m16*16+ln][c=kc*32+kval(pos)]
// ---------------------------------------------------------------------------
__global__ void build_fg_kernel(_Float16* __restrict__ Ft, _Float16* __restrict__ Gt) {
  int idx = blockIdx.x * blockDim.x + threadIdx.x;
  const float TWO_PI = 6.283185307179586f;
  if (idx < 64 * 256) {
    int pos = idx & 31, ln = (idx >> 5) & 15, kc = (idx >> 9) & 7, m16 = idx >> 12;
    int c = m16 * 16 + ln;                 // F row (0..31 re, 32..63 im)
    int t = kc * 32 + posOf(pos);          // time index
    int r = c & 31;
    float ang = TWO_PI * (float)((r * t) & 255) * (1.0f / 256.0f);
    float v = (c < 32) ? __cosf(ang) : -__sinf(ang);
    Ft[idx] = (_Float16)(v * (1.0f / 256.0f));
  } else if (idx < 2 * 64 * 256) {
    int j = idx - 64 * 256;
    int pos = j & 31, ln = (j >> 5) & 15, kc = (j >> 9) & 1, m16 = j >> 10;
    int x = m16 * 16 + ln;
    int c = kc * 32 + posOf(pos);
    int r = c & 31;
    float ang = TWO_PI * (float)((r * x) & 255) * (1.0f / 256.0f);
    float v;
    if (c < 32) v = (r == 0) ? 1.0f : 2.0f * __cosf(ang);
    else        v = (r == 0) ? 0.0f : -2.0f * __sinf(ang);
    Gt[j] = (_Float16)v;
  }
}

// ---------------------------------------------------------------------------
// Complex-mix operator, B-fragment-ready: Mt[r=32][kc=4][n=128][32]
//   value = Mbig[r][j2 = kc*32+kval(pos)][i2 = n]
//   Mbig[j2=p*64+j][i2=q*64+i]: q=0: p=0->kr, p=1->-ki ; q=1: p=0->ki, p=1->kr
// ---------------------------------------------------------------------------
__global__ void build_mix_kernel(const float* __restrict__ kr, const float* __restrict__ ki,
                                 _Float16* __restrict__ Mt) {
  int idx = blockIdx.x * blockDim.x + threadIdx.x;
  if (idx >= 32 * 4 * 128 * 32) return;
  int pos = idx & 31, n = (idx >> 5) & 127, kc = (idx >> 12) & 3, r = idx >> 14;
  int j2 = kc * 32 + posOf(pos), i2 = n;
  int p = j2 >> 6, j = j2 & 63, q = i2 >> 6, i = i2 & 63;
  long ko = (long)r * 4096 + (long)i * 64 + j;
  float v = (q == 0) ? ((p == 0) ? kr[ko] : -ki[ko])
                     : ((p == 0) ? ki[ko] :  kr[ko]);
  Mt[idx] = (_Float16)v;
}

// ---------------------------------------------------------------------------
// Generic batched WMMA GEMM: C[64*gy x 64*gx] = A[.. x K] @ B[K x ..]
// A is fragment-ready in global ([m16][K/32][16][32]); B staged through a
// K-permuted, transposed LDS tile so B fragments are 2 x ds_load_b128.
// block = 128 (4 waves); each wave owns one 16-row strip x 64 cols.
// ---------------------------------------------------------------------------
template <bool B_IS_F32, bool C_STORE_F16, bool C_ACCUM>
__global__ __launch_bounds__(128)
void wmma_gemm_kernel(const _Float16* __restrict__ At,
                      const void* __restrict__ Bv, long bO, long bI,
                      void* __restrict__ Cv, long cO, long cI,
                      int batchMod, int ldb, int ldc, int K) {
  __shared__ __align__(16) _Float16 Bt[64][40];   // [n][K-permuted], 80B row skew
  const int tid = threadIdx.x, lane = tid & 31, wave = tid >> 5;
  const int ln = lane & 15, khi = lane >> 4;
  const int outer = blockIdx.z / batchMod, inner = blockIdx.z % batchMod;
  const long boff = (long)outer * bO + (long)inner * bI;
  const long coff = (long)outer * cO + (long)inner * cI;
  const int mbase = blockIdx.y * 64, nbase = blockIdx.x * 64;
  const int Kc = K >> 5;

  v8f acc[4] = {};

  for (int kc = 0; kc < Kc; ++kc) {
    const int k0 = kc << 5;
    if (B_IS_F32) {
      // 32x64 f32 tile: 512 float4 chunks, convert + K-permuted scatter
#pragma unroll
      for (int it = 0; it < 4; ++it) {
        int cid = it * 128 + tid;                 // [0,512)
        int k = cid >> 4, n4 = (cid & 15) << 2;
        const float4 vsrc = *(const float4*)((const float*)Bv + boff +
                                             (long)(k0 + k) * ldb + (nbase + n4));
        int kp = posOf(k);
        Bt[n4 + 0][kp] = (_Float16)vsrc.x;
        Bt[n4 + 1][kp] = (_Float16)vsrc.y;
        Bt[n4 + 2][kp] = (_Float16)vsrc.z;
        Bt[n4 + 3][kp] = (_Float16)vsrc.w;
      }
    } else {
      // 32x64 f16 tile: 256 v8h chunks, K-permuted scatter
#pragma unroll
      for (int it = 0; it < 2; ++it) {
        int cid = it * 128 + tid;                 // [0,256)
        int k = cid >> 3, n8 = (cid & 7) << 3;
        const v8h vsrc = *(const v8h*)((const _Float16*)Bv + boff +
                                       (long)(k0 + k) * ldb + (nbase + n8));
        int kp = posOf(k);
#pragma unroll
        for (int u = 0; u < 8; ++u) Bt[n8 + u][kp] = vsrc[u];
      }
    }
    __syncthreads();
    // A fragment: direct from fragment-ready global (2 x global_load_b128)
    v16h a = load_frag16(At + (long)((blockIdx.y * 4 + wave) * Kc + kc) * 512 +
                         ln * 32 + khi * 16);
#pragma unroll
    for (int f = 0; f < 4; ++f) {
      v16h b = load_frag16(&Bt[f * 16 + ln][khi * 16]);   // 2 x ds_load_b128
      acc[f] = __builtin_amdgcn_wmma_f32_16x16x32_f16(
          false, a, false, b, (short)0, acc[f], false, false);
    }
    __syncthreads();
  }

#pragma unroll
  for (int f = 0; f < 4; ++f) {
#pragma unroll
    for (int v = 0; v < 8; ++v) {
      int row = mbase + wave * 16 + v + (khi << 3);       // C/D layout, ISA 7.12.2
      int col = nbase + f * 16 + ln;
      long off = coff + (long)row * ldc + col;
      if (C_STORE_F16) {
        ((_Float16*)Cv)[off] = (_Float16)acc[f][v];
      } else {
        float* Cf = (float*)Cv;
        if (C_ACCUM) Cf[off] += acc[f][v];
        else         Cf[off]  = acc[f][v];
      }
    }
  }
}

// ---------------------------------------------------------------------------
// Complex mix, per (b, mode r): Ycat[y,i2] = Xcat[y,j2] @ Mbig[r][j2,i2]
// A (Xf slabs) staged through K-permuted LDS; B fragments direct from Mt.
// grid = (4 y-tiles, 32 modes, 8 batch), block = 128.
// ---------------------------------------------------------------------------
__global__ __launch_bounds__(128)
void wmma_mix_kernel(const _Float16* __restrict__ Xf, const _Float16* __restrict__ Mt,
                     _Float16* __restrict__ Y) {
  __shared__ __align__(16) _Float16 As[64][40];
  const int tid = threadIdx.x, lane = tid & 31, wave = tid >> 5;
  const int ln = lane & 15, khi = lane >> 4;
  const int ybase = blockIdx.x * 64, r = blockIdx.y, b = blockIdx.z;
  const long xf_b = (long)b * (64L * S * Cc);

  v8f acc[8] = {};

  for (int kc = 0; kc < 4; ++kc) {
    const int k0 = kc << 5;
    const int p = k0 >> 6;                 // re/im half of contraction
    const int jb = k0 & 63;                // channel offset within half
    const long abase = xf_b + (long)(p * 32 + r) * (S * Cc);
    // 64x32 f16 tile, 8-half chunks: b128 load + b128 store (pos8 keeps groups)
#pragma unroll
    for (int it = 0; it < 2; ++it) {
      int cid = it * 128 + tid;            // [0,256)
      int row = cid >> 2, kk8 = (cid & 3) << 3;
      const v8h vsrc = *(const v8h*)(Xf + abase + (long)(ybase + row) * Cc + (jb + kk8));
      *(v8h*)&As[row][pos8(kk8)] = vsrc;
    }
    __syncthreads();
    v16h a = load_frag16(&As[wave * 16 + ln][khi * 16]);
#pragma unroll
    for (int f = 0; f < 8; ++f) {
      v16h bb = load_frag16(Mt + (long)((r * 4 + kc) * 128 + f * 16 + ln) * 32 + khi * 16);
      acc[f] = __builtin_amdgcn_wmma_f32_16x16x32_f16(
          false, a, false, bb, (short)0, acc[f], false, false);
    }
    __syncthreads();
  }

#pragma unroll
  for (int f = 0; f < 8; ++f) {
#pragma unroll
    for (int v = 0; v < 8; ++v) {
      int y = ybase + wave * 16 + v + (khi << 3);
      int i2 = f * 16 + ln;
      int q = i2 >> 6, i = i2 & 63;
      long off = (long)b * (64L * S * Cc) + (long)(q * 32 + r) * (S * Cc) + (long)y * Cc + i;
      Y[off] = (_Float16)acc[f][v];
    }
  }
}

// ---------------------------------------------------------------------------
// Host launcher
// ---------------------------------------------------------------------------
extern "C" void kernel_launch(void* const* d_in, const int* in_sizes, int n_in,
                              void* d_out, int out_size, void* d_ws, size_t ws_size,
                              hipStream_t stream) {
  const float* X   = (const float*)d_in[0];
  const float* k0r = (const float*)d_in[1];
  const float* k0i = (const float*)d_in[2];
  const float* k1r = (const float*)d_in[3];
  const float* k1i = (const float*)d_in[4];
  float* out = (float*)d_out;

  char* ws = (char*)d_ws;
  _Float16* Ft = (_Float16*)(ws);                               // 32 KB
  _Float16* Gt = (_Float16*)(ws + 32768);                       // 32 KB
  _Float16* M0 = (_Float16*)(ws + 65536);                       //  1 MB
  _Float16* M1 = (_Float16*)(ws + 65536 + 1048576);             //  1 MB
  _Float16* Xf = (_Float16*)(ws + 65536 + 2 * 1048576);         // 16 MB
  _Float16* Y  = (_Float16*)(ws + 65536 + 2 * 1048576 + 16777216);

  const long SXB = (long)S * S * Cc;   // per-b stride in X / out = 4194304
  const long XFB = 64L * S * Cc;       // per-b stride in Xf / Y  = 1048576
  const int  SC  = S * Cc;             // 16384

  build_fg_kernel<<<(2 * 64 * 256 + 255) / 256, 256, 0, stream>>>(Ft, Gt);
  build_mix_kernel<<<(32 * 128 * 128 + 255) / 256, 256, 0, stream>>>(k0r, k0i, M0);
  build_mix_kernel<<<(32 * 128 * 128 + 255) / 256, 256, 0, stream>>>(k1r, k1i, M1);

  // ---- branch 0: contract over x (axis -3) -------------------------------
  wmma_gemm_kernel<true, true, false><<<dim3(SC / 64, 1, Bn), 128, 0, stream>>>(
      Ft, X, SXB, 0, Xf, XFB, 0, 1, SC, SC, 256);
  wmma_mix_kernel<<<dim3(4, 32, Bn), 128, 0, stream>>>(Xf, M0, Y);
  wmma_gemm_kernel<false, false, false><<<dim3(SC / 64, 4, Bn), 128, 0, stream>>>(
      Gt, Y, XFB, 0, out, SXB, 0, 1, SC, SC, 64);

  // ---- branch 1: contract over y (axis -2) -------------------------------
  wmma_gemm_kernel<true, true, false><<<dim3(1, 1, Bn * S), 128, 0, stream>>>(
      Ft, X, SXB, (long)SC, Xf, XFB, (long)Cc, S, Cc, SC, 256);
  wmma_mix_kernel<<<dim3(4, 32, Bn), 128, 0, stream>>>(Xf, M1, Y);
  wmma_gemm_kernel<false, false, true><<<dim3(1, 4, Bn * S), 128, 0, stream>>>(
      Gt, Y, XFB, (long)Cc, out, SXB, (long)SC, S, SC, Cc, 64);
}
